// SoftMOE_12919261626414
// MI455X (gfx1250) — matrix operand order repaired and checked by reading, not compiled, not run on hardware
//
#include <hip/hip_runtime.h>
#include <hip/hip_bf16.h>

// Problem dims (fixed by the reference)
#define T_TOK 8192
#define DIM   1024
#define NEXP  64
#define NP    128   // experts * slots
#define EDIM  1024

#define AS1 __attribute__((address_space(1)))
#define AS3 __attribute__((address_space(3)))

typedef __bf16 bf16_t;
typedef __attribute__((ext_vector_type(16))) bf16_t v16bf;
typedef __attribute__((ext_vector_type(8)))  bf16_t v8bf;
typedef __attribute__((ext_vector_type(8)))  float  v8f;
typedef __attribute__((ext_vector_type(4)))  int    v4i;

union AFrag { v16bf v; v8bf h[2]; };

// float -> bf16, round-to-nearest-even via bit math (no libcall risk)
__device__ __forceinline__ bf16_t f2bf(float f) {
    unsigned u = __builtin_bit_cast(unsigned, f);
    unsigned r = u + 0x7FFFu + ((u >> 16) & 1u);
    unsigned short h = (unsigned short)(r >> 16);
    return __builtin_bit_cast(bf16_t, h);
}

// ---------------------------------------------------------------------------
// CDNA5 async copy: GLOBAL_LOAD_ASYNC_TO_LDS_B128 (16 bytes), ASYNCcnt-tracked.
// Builtin signature (probed round 2): takes pointers to 16-byte int vectors.
// ---------------------------------------------------------------------------
__device__ __forceinline__ void async_b128(const bf16_t* g, bf16_t* l) {
#if __has_builtin(__builtin_amdgcn_global_load_async_to_lds_b128)
    __builtin_amdgcn_global_load_async_to_lds_b128(
        (AS1 v4i*)(AS1 void*)(void*)const_cast<bf16_t*>(g),
        (AS3 v4i*)(AS3 void*)(void*)l, 0, 0);
#else
    unsigned loff = (unsigned)(unsigned long long)(const AS3 bf16_t*)(const AS3 void*)(const void*)l;
    asm volatile("global_load_async_to_lds_b128 %0, %1, off"
                 :: "v"(loff), "v"(g) : "memory");
#endif
}

__device__ __forceinline__ void wait_async0() {
#if __has_builtin(__builtin_amdgcn_s_wait_asynccnt)
    __builtin_amdgcn_s_wait_asynccnt(0);
#else
    asm volatile("s_wait_asynccnt 0" ::: "memory");
#endif
}

// Force an LDS (AS3) read -> ds_load_b128 x2
__device__ __forceinline__ v16bf lds_read16(const bf16_t* p) {
    return *(const AS3 v16bf*)(const AS3 void*)(const void*)p;
}

// ---------------------------------------------------------------------------
// Block-level GEMM: 128 threads = 4 waves; block computes a 16-row x 64-col
// f32 tile of C = A(bf16 row-major) * B(bf16 column-major: Bcm[col*ldb + k]).
// B chunks (64 cols x 32 K = 4 KB) are double-buffered in LDS via
// global_load_async_to_lds_b128 + s_wait_asynccnt; A fragments are direct
// b128 global loads. v_wmma_f32_16x16x32_bf16 per 32-K step.
// A and B fragments both use contiguous K order k = 16h + s (wave32).
// ---------------------------------------------------------------------------
__device__ __forceinline__ void gemm_block_bf16(
    const bf16_t* __restrict__ A,   size_t lda,
    const bf16_t* __restrict__ Bcm, size_t ldb,
    float*        __restrict__ C,   size_t ldc,
    int K, int row0, int colBase)
{
    __shared__ __align__(64) bf16_t sB[2 * 64 * 32];   // 8 KB, double buffered

    const int tid  = threadIdx.x;
    const int w    = tid >> 5;       // wave id: owns cols [colBase + 16w, +16)
    const int lane = tid & 31;
    const int half = lane >> 4;      // 0: lanes 0-15, 1: lanes 16-31
    const int l16  = lane & 15;

    // Copy role: thread t loads 32 bytes of column (tid>>1), K-half (tid&1)
    const int ccol = tid >> 1;
    const int ch   = tid & 1;
    const bf16_t* gcol = Bcm + (size_t)(colBase + ccol) * ldb + ch * 16;
    const int lbase = ccol * 32 + ch * 16;

    // Prologue: chunk 0 -> buffer 0
    async_b128(gcol,     &sB[lbase]);
    async_b128(gcol + 8, &sB[lbase + 8]);
    wait_async0();
    __syncthreads();

    const bf16_t* arow = A + (size_t)(row0 + l16) * lda;

    v8f acc = {};
    const int nchunk = K >> 5;
    for (int ck = 0; ck < nchunk; ++ck) {
        const int cur = ck & 1;

        // Prefetch next chunk into the other buffer (stale, safe to overwrite)
        if (ck + 1 < nchunk) {
            const bf16_t* g = gcol + (size_t)(ck + 1) * 32;
            bf16_t* l = &sB[(1 - cur) * 2048 + lbase];
            async_b128(g,     l);
            async_b128(g + 8, l + 8);
        }

        // A fragment: lane holds row l16, 16 contiguous K at [32ck+16h..+15]
        const bf16_t* ap = arow + ck * 32 + half * 16;
        AFrag a;
        a.h[0] = *(const v8bf*)ap;
        a.h[1] = *(const v8bf*)(ap + 8);

        // B fragment from LDS: lane holds col (16w + l16), 16 contiguous K at 16h
        v16bf b = lds_read16(&sB[cur * 2048 + (w * 16 + l16) * 32 + half * 16]);

        acc = __builtin_amdgcn_wmma_f32_16x16x32_bf16(
            false, a.v, false, b, (short)0, acc, false, false);

        wait_async0();      // this wave's async writes for next chunk landed
        __syncthreads();    // everyone done reading 'cur' + next chunk visible
    }

    // C layout: VGPR r -> row (r + 8h), col l16
    float* crow = C + (size_t)(row0 + half * 8) * ldc + (colBase + w * 16 + l16);
#pragma unroll
    for (int r = 0; r < 8; ++r) crow[(size_t)r * ldc] = acc[r];
}

// ---------------------------------------------------------------------------
// Kernel 0: tiled transpose + f32->bf16. src [R,C] f32 -> dstT [C,R] bf16,
// and optionally a row-major bf16 copy dstRM [R,C]. block (32,8), grid (C/32,R/32).
// ---------------------------------------------------------------------------
__global__ void k_transpose_cvt(const float* __restrict__ src,
                                bf16_t* __restrict__ dstT,
                                bf16_t* __restrict__ dstRM,
                                int R, int C)
{
    __shared__ float tile[32][33];
    const int c  = blockIdx.x * 32 + threadIdx.x;
    const int r0 = blockIdx.y * 32;
#pragma unroll
    for (int j = threadIdx.y; j < 32; j += 8) {
        const float v = src[(size_t)(r0 + j) * C + c];
        tile[j][threadIdx.x] = v;
        if (dstRM) dstRM[(size_t)(r0 + j) * C + c] = f2bf(v);
    }
    __syncthreads();
    const int rr = r0 + threadIdx.x;
    const int cb = blockIdx.x * 32;
#pragma unroll
    for (int j = threadIdx.y; j < 32; j += 8)
        dstT[(size_t)(cb + j) * R + rr] = f2bf(tile[threadIdx.x][j]);
}

// ---------------------------------------------------------------------------
// Kernel 1: logits[T,128] = tokens[T,D] @ phi[D,128]
//   A = tokens bf16 [T,D], B = phiT bf16 [128,D]. grid (T/16, NP/64), block 128.
// ---------------------------------------------------------------------------
__global__ void k_logits(const bf16_t* __restrict__ tokens_bf,
                         const bf16_t* __restrict__ phiT,
                         float* __restrict__ logits)
{
    gemm_block_bf16(tokens_bf, DIM, phiT, DIM, logits, NP,
                    DIM, blockIdx.x * 16, blockIdx.y * 64);
}

// ---------------------------------------------------------------------------
// Kernel 2: per-token softmaxes over 128 logits.
//   dispatch: softmax over experts n per slot p (parity classes)
//   combine : joint softmax over all 128
// Emits bf16 dispatchT [128,T] and bf16 combine [T,128] (GEMM A operands).
// ---------------------------------------------------------------------------
__global__ void k_softmax(const float* __restrict__ logits,
                          bf16_t* __restrict__ dispatchT,
                          bf16_t* __restrict__ combine)
{
    const int t = blockIdx.x;
    const int i = threadIdx.x;          // i = n*2 + p
    __shared__ float sm[NP];

    const float l = logits[(size_t)t * NP + i];
    sm[i] = l; __syncthreads();
#pragma unroll
    for (int s = 64; s >= 1; s >>= 1) { if (i < s) sm[i] = fmaxf(sm[i], sm[i + s]); __syncthreads(); }
    const float M = sm[0];
    __syncthreads();

    const float e = __expf(l - M);
    sm[i] = e; __syncthreads();
    // stride stays even -> parity (slot p) preserved; sm[0]=sum(p=0), sm[1]=sum(p=1)
#pragma unroll
    for (int s = 64; s >= 2; s >>= 1) { if (i < s) sm[i] += sm[i + s]; __syncthreads(); }

    const float denom_p = (i & 1) ? sm[1] : sm[0];
    const float denom_j = sm[0] + sm[1];

    dispatchT[(size_t)i * T_TOK + t] = f2bf(e / denom_p);
    combine  [(size_t)t * NP + i]    = f2bf(e / denom_j);
}

// ---------------------------------------------------------------------------
// Kernel 3: mixture[128,D] = dispatchT[128,T] @ tokens[T,D]
//   A = dispatchT bf16 [128,T], B = tokensT bf16 [D,T]. grid (NP/16, D/64).
// ---------------------------------------------------------------------------
__global__ void k_mixture(const bf16_t* __restrict__ dispatchT,
                          const bf16_t* __restrict__ tokensT,
                          float* __restrict__ mixture)
{
    gemm_block_bf16(dispatchT, T_TOK, tokensT, T_TOK, mixture, DIM,
                    T_TOK, blockIdx.x * 16, blockIdx.y * 64);
}

// ---------------------------------------------------------------------------
// Kernel 4: per-expert dense + ReLU — the single pass over W (256 MB, HBM
// roofline term). float4 W loads, 4 outputs/thread, LDS-cached mixture rows,
// global_prefetch_b8 ahead of the stream. Writes Ys transposed+bf16:
// YsT[e*128 + 2n+p] (stage-5 B layout). grid (NEXP), block 256.
// ---------------------------------------------------------------------------
__global__ void k_ffn(const float* __restrict__ mixture, const float* __restrict__ W,
                      const float* __restrict__ b, bf16_t* __restrict__ YsT)
{
    const int n  = blockIdx.x;
    const int e0 = threadIdx.x * 4;

    __shared__ float m0[DIM], m1[DIM];
    for (int d = threadIdx.x; d < DIM; d += 256) {
        m0[d] = mixture[(size_t)(2 * n)     * DIM + d];
        m1[d] = mixture[(size_t)(2 * n + 1) * DIM + d];
    }
    __syncthreads();

    const float* Wp = W + (size_t)n * DIM * EDIM + e0;
    float acc0[4] = {0.f, 0.f, 0.f, 0.f};
    float acc1[4] = {0.f, 0.f, 0.f, 0.f};
#pragma unroll 4
    for (int d = 0; d < DIM; ++d) {
        if ((d & 31) == 0 && d + 32 < DIM)
            __builtin_prefetch(Wp + (size_t)(d + 32) * EDIM, 0, 0);  // global_prefetch_b8
        const float4 wv = *(const float4*)(Wp + (size_t)d * EDIM);
        const float a = m0[d], c = m1[d];
        acc0[0] = fmaf(a, wv.x, acc0[0]); acc0[1] = fmaf(a, wv.y, acc0[1]);
        acc0[2] = fmaf(a, wv.z, acc0[2]); acc0[3] = fmaf(a, wv.w, acc0[3]);
        acc1[0] = fmaf(c, wv.x, acc1[0]); acc1[1] = fmaf(c, wv.y, acc1[1]);
        acc1[2] = fmaf(c, wv.z, acc1[2]); acc1[3] = fmaf(c, wv.w, acc1[3]);
    }
#pragma unroll
    for (int j = 0; j < 4; ++j) {
        const float bb = b[(size_t)n * EDIM + e0 + j];
        YsT[(size_t)(e0 + j) * NP + 2 * n]     = f2bf(fmaxf(acc0[j] + bb, 0.f));
        YsT[(size_t)(e0 + j) * NP + 2 * n + 1] = f2bf(fmaxf(acc1[j] + bb, 0.f));
    }
}

// ---------------------------------------------------------------------------
// Kernel 5: Y[T,E] = combine[T,128] @ Ys[128,E]
//   A = combine bf16 [T,128], B = YsT bf16 [E,128]. grid (T/16, E/64).
// ---------------------------------------------------------------------------
__global__ void k_combine_out(const bf16_t* __restrict__ combine,
                              const bf16_t* __restrict__ YsT,
                              float* __restrict__ Y)
{
    gemm_block_bf16(combine, NP, YsT, NP, Y, EDIM,
                    NP, blockIdx.x * 16, blockIdx.y * 64);
}

// ---------------------------------------------------------------------------
extern "C" void kernel_launch(void* const* d_in, const int* in_sizes, int n_in,
                              void* d_out, int out_size, void* d_ws, size_t ws_size,
                              hipStream_t stream)
{
    const float* tokens = (const float*)d_in[0];   // [T, D]
    const float* phi    = (const float*)d_in[1];   // [D, N, P] = [D, 128]
    const float* W      = (const float*)d_in[2];   // [N, D, E]
    const float* b      = (const float*)d_in[3];   // [N, E]
    float*       Y      = (float*)d_out;           // [T, E]

    // Workspace carve-up (~41 MB, all chunks 256B-aligned by construction)
    char* p = (char*)d_ws;
    bf16_t* tokensT   = (bf16_t*)p; p += (size_t)DIM * T_TOK * sizeof(bf16_t); // [D, T]
    bf16_t* tokens_bf = (bf16_t*)p; p += (size_t)T_TOK * DIM * sizeof(bf16_t); // [T, D]
    bf16_t* phiT      = (bf16_t*)p; p += (size_t)NP  * DIM   * sizeof(bf16_t); // [128, D]
    float*  logits    = (float*)p;  p += (size_t)T_TOK * NP  * sizeof(float);  // [T, 128]
    bf16_t* dispatchT = (bf16_t*)p; p += (size_t)NP * T_TOK  * sizeof(bf16_t); // [128, T]
    bf16_t* combine   = (bf16_t*)p; p += (size_t)T_TOK * NP  * sizeof(bf16_t); // [T, 128]
    float*  mixture   = (float*)p;  p += (size_t)NP * DIM    * sizeof(float);  // [128, D]
    bf16_t* YsT       = (bf16_t*)p; p += (size_t)EDIM * NP   * sizeof(bf16_t); // [E, 128]

    // Stage 0: bf16 staging (transposed for B operands, row-major tokens for A)
    k_transpose_cvt<<<dim3(DIM / 32, T_TOK / 32), dim3(32, 8), 0, stream>>>(
        tokens, tokensT, tokens_bf, T_TOK, DIM);
    k_transpose_cvt<<<dim3(NP / 32, DIM / 32), dim3(32, 8), 0, stream>>>(
        phi, phiT, (bf16_t*)nullptr, DIM, NP);

    // Stage 1: logits
    k_logits<<<dim3(T_TOK / 16, NP / 64), 128, 0, stream>>>(tokens_bf, phiT, logits);

    // Stage 2: softmaxes
    k_softmax<<<dim3(T_TOK), 128, 0, stream>>>(logits, dispatchT, combine);

    // Stage 3: mixture inputs
    k_mixture<<<dim3(NP / 16, DIM / 64), 128, 0, stream>>>(dispatchT, tokensT, mixture);

    // Stage 4: expert dense + ReLU (the W pass)
    k_ffn<<<dim3(NEXP), 256, 0, stream>>>(mixture, W, b, YsT);

    // Stage 5: combine back to tokens
    k_combine_out<<<dim3(T_TOK / 16, EDIM / 64), 128, 0, stream>>>(combine, YsT, Y);
}
